// TinyStoriesInt8Surrogate_76295799046220
// MI455X (gfx1250) — 8-bit matrix operand order repaired and checked by reading, compiled, not run on hardware
//
#include <hip/hip_runtime.h>
#include <hip/hip_bf16.h>

typedef __attribute__((ext_vector_type(8))) int v8i;

#define VOCAB 50257
#define VPAD  50304
#define D     768
#define ROWS  4096
#define SEQ   2048
#define NTILE 64
#define LDS_STRIDE 784

__global__ __launch_bounds__(256) void pack_lm_head(const int* __restrict__ w,
                                                    signed char* __restrict__ bp) {
  __shared__ int tile[64][17];
  const int t  = threadIdx.x;
  const int n0 = blockIdx.x * 64;
  const int k0 = blockIdx.y * 64;

  for (int i = 0; i < 16; ++i) {
    const int e  = t + i * 256;
    const int k  = e >> 6;
    const int nn = e & 63;
    const int gn = n0 + nn;
    const int val = (gn < VOCAB) ? w[(size_t)(k0 + k) * VOCAB + gn] : 0;
    ((signed char*)&tile[nn][0])[k] = (signed char)val;
  }
  __syncthreads();

  const int nn = t >> 2;
  const int c  = (t & 3) * 4;
  int4 v;
  v.x = tile[nn][c + 0];
  v.y = tile[nn][c + 1];
  v.z = tile[nn][c + 2];
  v.w = tile[nn][c + 3];
  *(int4*)(bp + (size_t)(n0 + nn) * D + k0 + c * 4) = v;
}

__global__ __launch_bounds__(256) void int8_logits_gemm(
    const int*         __restrict__ ids,
    const signed char* __restrict__ wte,
    const signed char* __restrict__ wpe,
    const signed char* __restrict__ bp,
    const int*         __restrict__ bias,
    float*             __restrict__ out)
{
  __shared__ __align__(16) signed char lb[NTILE * LDS_STRIDE];

  const int t  = threadIdx.x;
  const int n0 = blockIdx.x * NTILE;

  for (int idx = t; idx < NTILE * (D / 16); idx += 256) {
    const int c  = idx / (D / 16);
    const int kk = (idx % (D / 16)) * 16;
    const signed char* gsrc = bp + (size_t)(n0 + c) * D + kk;
    unsigned ldsoff = (unsigned)(size_t)(&lb[c * LDS_STRIDE + kk]);
    unsigned long long ga = (unsigned long long)(size_t)gsrc;
    asm volatile("global_load_async_to_lds_b128 %0, %1, off"
                 :: "v"(ldsoff), "v"(ga) : "memory");
  }
  asm volatile("s_wait_asynccnt 0" ::: "memory");
  __syncthreads();

  const int lane  = t & 31;
  const int wave  = t >> 5;
  const int cg    = wave & 3;
  const int mh    = wave >> 2;
  const int n     = lane & 15;
  const int koffA = (lane < 16) ? 0 : 16;
  const int koffB = (lane < 16) ? 0 : 16;

  const signed char* lbcol = &lb[(cg * 16 + n) * LDS_STRIDE + koffB];
  const int col = n0 + cg * 16 + n;
  const int bv  = (col < VOCAB) ? bias[col] : 0;

  for (int m0 = mh * 16; m0 < ROWS; m0 += 32) {
    const int row = m0 + n;
    const int tok = ids[row];
    const int pos = row & (SEQ - 1);
    const signed char* at = wte + (size_t)tok * D + koffA;
    const signed char* ap = wpe + (size_t)pos * D + koffA;

    v8i accT = {};
    v8i accP = {};

#pragma unroll 4
    for (int k0 = 0; k0 < D; k0 += 64) {
      int2 t0 = *(const int2*)(at + k0);
      int2 t1 = *(const int2*)(at + k0 + 8);
      int2 t2 = *(const int2*)(at + k0 + 32);
      int2 t3 = *(const int2*)(at + k0 + 40);
      v8i aT = (v8i){t0.x, t0.y, t1.x, t1.y, t2.x, t2.y, t3.x, t3.y};

      int2 p0 = *(const int2*)(ap + k0);
      int2 p1 = *(const int2*)(ap + k0 + 8);
      int2 p2 = *(const int2*)(ap + k0 + 32);
      int2 p3 = *(const int2*)(ap + k0 + 40);
      v8i aP = (v8i){p0.x, p0.y, p1.x, p1.y, p2.x, p2.y, p3.x, p3.y};

      int4 q0 = *(const int4*)(lbcol + k0);
      int4 q1 = *(const int4*)(lbcol + k0 + 32);
      v8i b = (v8i){q0.x, q0.y, q0.z, q0.w, q1.x, q1.y, q1.z, q1.w};

      accT = __builtin_amdgcn_wmma_i32_16x16x64_iu8(true, aT, true, b, accT, false, false);
      accP = __builtin_amdgcn_wmma_i32_16x16x64_iu8(true, aP, true, b, accP, false, false);
    }

    if (col < VOCAB) {
      const int mbase = m0 + ((lane < 16) ? 0 : 8);
#pragma unroll
      for (int v = 0; v < 8; ++v)
        out[(size_t)(mbase + v) * VOCAB + col] = (float)(accT[v] + accP[v] + bv);
    }
  }
}

extern "C" void kernel_launch(void* const* d_in, const int* in_sizes, int n_in,
                              void* d_out, int out_size, void* d_ws, size_t ws_size,
                              hipStream_t stream) {
  const int*         ids  = (const int*)d_in[0];
  const signed char* wte  = (const signed char*)d_in[1];
  const signed char* wpe  = (const signed char*)d_in[2];
  const int*         lm   = (const int*)d_in[3];
  const int*         bias = (const int*)d_in[4];
  float*             out  = (float*)d_out;
  signed char*       bp   = (signed char*)d_ws;

  dim3 pgrid(VPAD / 64, D / 64);
  pack_lm_head<<<pgrid, 256, 0, stream>>>(lm, bp);

  dim3 ggrid(VPAD / NTILE);
  int8_logits_gemm<<<ggrid, 256, 0, stream>>>(ids, wte, wpe, bp, bias, out);
}
